// SLSTMWoDummy_11562051961228
// MI455X (gfx1250) — matrix operand order repaired and checked by reading, not compiled, not run on hardware
//
#include <hip/hip_runtime.h>

// ---------------------------------------------------------------------------
// Types for CDNA5 WMMA
// ---------------------------------------------------------------------------
typedef __attribute__((ext_vector_type(16))) __bf16 v16bf;
typedef __attribute__((ext_vector_type(8)))  float  v8f;

union Frag {
    v16bf v;
    uint4 q[2];
};

__device__ __forceinline__ float sigmoidf_(float x) {
    return 1.0f / (1.0f + __expf(-x));
}

// ---------------------------------------------------------------------------
// Weight transpose + fp32 -> bf16 convert:  dstT[n*ldT + kofs + k] = src[k*N+n]
// ---------------------------------------------------------------------------
__global__ void transpose_convert(const float* __restrict__ src,
                                  __bf16* __restrict__ dst,
                                  int K, int N, int ldT, int kofs) {
    int idx = blockIdx.x * 256 + threadIdx.x;
    if (idx >= K * N) return;
    int k = idx / N;
    int n = idx - k * N;
    dst[(size_t)n * ldT + kofs + k] = (__bf16)src[(size_t)k * N + n];
}

// ---------------------------------------------------------------------------
// x = word_inputs * (1-mask); h = c = x; cat block3 (constant across layers)
// ---------------------------------------------------------------------------
__global__ void init_state(const float* __restrict__ wi,
                           const int* __restrict__ mask,
                           float* __restrict__ h, float* __restrict__ c,
                           __bf16* __restrict__ cat) {
    int idx = blockIdx.x * 256 + threadIdx.x;   // over 8192*512
    int row = idx >> 9;
    int e   = idx & 511;
    float sm = 1.0f - (float)mask[row];
    float x  = wi[idx] * sm;
    h[idx] = x;
    c[idx] = x;
    cat[(size_t)row * 2048 + 1536 + e] = (__bf16)x;
}

// ---------------------------------------------------------------------------
// cat[:,0:512]=h  cat[:,512:1024]=hb  cat[:,1024:1536]=ha   (bf16)
// ---------------------------------------------------------------------------
__global__ void build_cat(const float* __restrict__ h,
                          const int* __restrict__ mask,
                          __bf16* __restrict__ cat) {
    int idx = blockIdx.x * 256 + threadIdx.x;   // over 8192*512
    int row = idx >> 9;
    int e   = idx & 511;
    int s   = row & 255;                        // S = 256
    float sm = 1.0f - (float)mask[row];
    float hc = h[idx];
    float hb = 0.0f, ha = 0.0f;
    if (s >= 1)   hb += h[idx - 512];
    if (s >= 2)   hb += h[idx - 1024];
    if (s <= 254) ha += h[idx + 512];
    if (s <= 253) ha += h[idx + 1024];
    hb *= sm;
    ha *= sm;
    size_t base = (size_t)row * 2048;
    cat[base + e]        = (__bf16)hc;
    cat[base + 512 + e]  = (__bf16)hb;
    cat[base + 1024 + e] = (__bf16)ha;
}

// ---------------------------------------------------------------------------
// bf16 WMMA GEMM:  OUT[m,n] = act( sum_k A[m,k]*BT[n,k] + bias[n] + addend )
// Block tile 128x256, 8 waves of 64x64 (4x4 WMMA tiles), BK = 32, LDS staged.
// 16 WMMAs per 16 ds_load_b128 per wave per K-step (4x reuse of each frag).
// act: 0=none, 1=relu, 2=tanh. outF (fp32) and/or outH (bf16) stores.
// ---------------------------------------------------------------------------
__global__ void __launch_bounds__(256)
gemm_bf16(const __bf16* __restrict__ A, int lda,
          const __bf16* __restrict__ BT, int ldb,
          const float* __restrict__ bias,
          const float* __restrict__ addend, int ldadd,
          float* __restrict__ outF, __bf16* __restrict__ outH, int ldc,
          int K, int act) {
    __shared__ alignas(16) __bf16 sA[128][40];   // padded stride: 80B rows
    __shared__ alignas(16) __bf16 sB[256][40];

    const int tid  = threadIdx.x;
    const int m0   = blockIdx.y * 128;
    const int n0   = blockIdx.x * 256;
    const int arow = tid >> 1;                   // 0..127 (A tile row)
    const int acol = (tid & 1) * 16;             // 0 or 16
    const int lane = tid & 31;
    const int w    = tid >> 5;                   // 8 waves
    const int wm   = w >> 2;                     // 0..1   (M: 64 rows each)
    const int wn   = w & 3;                      // 0..3   (N: 64 cols each)
    const int nlo  = lane & 15;
    const int khalf = lane >> 4;

    v8f acc[4][4] = {};

    const __bf16* ag  = A  + (size_t)(m0 + arow) * lda + acol;
    const __bf16* bgp = BT + (size_t)(n0 + tid) * ldb;   // one B row per thread

    for (int k0 = 0; k0 < K; k0 += 32) {
        uint4 a0 = *(const uint4*)(ag + k0);
        uint4 a1 = *(const uint4*)(ag + k0 + 8);
        uint4 b0 = *(const uint4*)(bgp + k0);
        uint4 b1 = *(const uint4*)(bgp + k0 + 8);
        uint4 b2 = *(const uint4*)(bgp + k0 + 16);
        uint4 b3 = *(const uint4*)(bgp + k0 + 24);
        if (k0 + 32 < K) {
            __builtin_prefetch(ag + k0 + 32, 0, 3);   // global_prefetch_b8
            __builtin_prefetch(bgp + k0 + 32, 0, 3);
        }
        __syncthreads();
        *(uint4*)(&sA[arow][acol])     = a0;
        *(uint4*)(&sA[arow][acol + 8]) = a1;
        *(uint4*)(&sB[tid][0])         = b0;
        *(uint4*)(&sB[tid][8])         = b1;
        *(uint4*)(&sB[tid][16])        = b2;
        *(uint4*)(&sB[tid][24])        = b3;
        __syncthreads();

        // Hold 4 B fragments: lane -> column nlo, 16 contiguous K at khalf*16
        Frag bf[4];
#pragma unroll
        for (int j = 0; j < 4; ++j) {
            const __bf16* p = &sB[wn * 64 + j * 16 + nlo][khalf * 16];
            bf[j].q[0] = *(const uint4*)(p);
            bf[j].q[1] = *(const uint4*)(p + 8);
        }
        // Stream A fragments: lane -> row nlo, 16 contiguous K at khalf*16
#pragma unroll
        for (int i = 0; i < 4; ++i) {
            Frag af;
            const __bf16* p = &sA[wm * 64 + i * 16 + nlo][khalf * 16];
            af.q[0] = *(const uint4*)(p);
            af.q[1] = *(const uint4*)(p + 8);
#pragma unroll
            for (int j = 0; j < 4; ++j) {
                acc[i][j] = __builtin_amdgcn_wmma_f32_16x16x32_bf16(
                    false, af.v, false, bf[j].v, (short)0, acc[i][j],
                    false, false);
            }
        }
    }

    // Epilogue: C/D layout: n = nlo, m = khalf*8 + vgpr_index
#pragma unroll
    for (int i = 0; i < 4; ++i) {
#pragma unroll
        for (int j = 0; j < 4; ++j) {
            int n = n0 + wn * 64 + j * 16 + nlo;
            float bv = bias ? bias[n] : 0.0f;
#pragma unroll
            for (int vi = 0; vi < 8; ++vi) {
                int m = m0 + wm * 64 + i * 16 + khalf * 8 + vi;
                float v = acc[i][j][vi] + bv;
                if (addend) v += addend[(size_t)m * ldadd + n];
                if (act == 1)      v = fmaxf(v, 0.0f);
                else if (act == 2) v = tanhf(v);
                if (outF) outF[(size_t)m * ldc + n] = v;
                if (outH) outH[(size_t)m * ldc + n] = (__bf16)v;
            }
        }
    }
}

// ---------------------------------------------------------------------------
// Per-row cell update: 6 fused LayerNorms + gates + softmax + c/h update.
// One block (256 threads) per (b,s) row; each thread owns elements tid, tid+256.
// ---------------------------------------------------------------------------
__global__ void __launch_bounds__(256)
cell_update(const float* __restrict__ gates, const float* __restrict__ emb,
            const float* __restrict__ cold, float* __restrict__ cnew,
            float* __restrict__ h, const int* __restrict__ mask,
            const float* __restrict__ gamma, const float* __restrict__ beta) {
    const int row = blockIdx.x;
    const int tid = threadIdx.x;
    const int s   = row & 255;
    const size_t gbase = (size_t)row * 3584;
    const float sm = 1.0f - (float)mask[row];

    const int gblk[6] = {0, 1, 2, 3, 4, 6};   // fl, fr, fc, i, e, o gate blocks

    __shared__ float sred[12][256];
    float gv[6][2];
#pragma unroll
    for (int j = 0; j < 6; ++j) {
        float a = gates[gbase + gblk[j] * 512 + tid];
        float b = gates[gbase + gblk[j] * 512 + tid + 256];
        gv[j][0] = a;
        gv[j][1] = b;
        sred[2 * j][tid]     = a + b;
        sred[2 * j + 1][tid] = a * a + b * b;
    }
    __syncthreads();
    for (int st = 128; st > 0; st >>= 1) {
        if (tid < st) {
#pragma unroll
            for (int q = 0; q < 12; ++q) sred[q][tid] += sred[q][tid + st];
        }
        __syncthreads();
    }

    float mean[6], rstd[6];
#pragma unroll
    for (int j = 0; j < 6; ++j) {
        float m  = sred[2 * j][0] * (1.0f / 512.0f);
        float vv = sred[2 * j + 1][0] * (1.0f / 512.0f) - m * m;
        mean[j] = m;
        rstd[j] = rsqrtf(vv + 1e-5f);
    }

#pragma unroll
    for (int k = 0; k < 2; ++k) {
        int e = tid + k * 256;
        float n[6];
#pragma unroll
        for (int j = 0; j < 6; ++j) {
            n[j] = (gv[j][k] - mean[j]) * rstd[j] * gamma[(size_t)j * 512 + e]
                   + beta[(size_t)j * 512 + e];
        }
        float gi = sigmoidf_(n[3]);
        float ge = sigmoidf_(n[4]);
        float go = sigmoidf_(n[5]);
        float cw = gates[gbase + 5 * 512 + e];
        float gw = tanhf(cw) + ge * emb[(size_t)row * 512 + e];

        float mx = fmaxf(n[0], fmaxf(n[1], n[2]));
        float e0 = __expf(n[0] - mx);
        float e1 = __expf(n[1] - mx);
        float e2 = __expf(n[2] - mx);
        float inv = 1.0f / (e0 + e1 + e2);

        size_t ci = (size_t)row * 512 + e;
        float cb = 0.0f, ca = 0.0f;
        if (s >= 1)   cb += cold[ci - 512];
        if (s >= 2)   cb += cold[ci - 1024];
        if (s <= 254) ca += cold[ci + 512];
        if (s <= 253) ca += cold[ci + 1024];
        cb *= sm;
        ca *= sm;

        float lc = (cb * e0 + ca * e1 + cold[ci] * e2) * inv;
        float cn = (lc * (1.0f - gi) + gw * gi) * sm;
        cnew[ci] = cn;
        h[ci] = go * tanhf(cn) * sm;
    }
}

// ---------------------------------------------------------------------------
// hn = LN(h) with gamma[6]/beta[6]; store fp32 + bf16 copies.
// ---------------------------------------------------------------------------
__global__ void __launch_bounds__(256)
final_ln(const float* __restrict__ h, const float* __restrict__ gamma,
         const float* __restrict__ beta, float* __restrict__ hn,
         __bf16* __restrict__ hnb) {
    const int row = blockIdx.x;
    const int tid = threadIdx.x;
    size_t base = (size_t)row * 512;
    float a = h[base + tid];
    float b = h[base + tid + 256];

    __shared__ float sred[2][256];
    sred[0][tid] = a + b;
    sred[1][tid] = a * a + b * b;
    __syncthreads();
    for (int st = 128; st > 0; st >>= 1) {
        if (tid < st) {
            sred[0][tid] += sred[0][tid + st];
            sred[1][tid] += sred[1][tid + st];
        }
        __syncthreads();
    }
    float m  = sred[0][0] * (1.0f / 512.0f);
    float vv = sred[1][0] * (1.0f / 512.0f) - m * m;
    float rs = rsqrtf(vv + 1e-5f);

#pragma unroll
    for (int k = 0; k < 2; ++k) {
        int e = tid + k * 256;
        float x = (k == 0) ? a : b;
        float v = (x - m) * rs * gamma[6 * 512 + e] + beta[6 * 512 + e];
        hn[base + e]  = v;
        hnb[base + e] = (__bf16)v;
    }
}

// ---------------------------------------------------------------------------
// Host orchestration
// ---------------------------------------------------------------------------
extern "C" void kernel_launch(void* const* d_in, const int* in_sizes, int n_in,
                              void* d_out, int out_size, void* d_ws, size_t ws_size,
                              hipStream_t stream) {
    (void)in_sizes; (void)n_in; (void)out_size; (void)ws_size;
    const float* wi       = (const float*)d_in[0];
    const int*   mask     = (const int*)d_in[1];
    const float* Wg       = (const float*)d_in[2];
    const float* bg       = (const float*)d_in[3];
    const float* We_gate  = (const float*)d_in[4];
    const float* We_trans = (const float*)d_in[5];
    const float* be_trans = (const float*)d_in[6];
    const float* W1       = (const float*)d_in[7];
    const float* b1       = (const float*)d_in[8];
    const float* W2       = (const float*)d_in[9];
    const float* b2       = (const float*)d_in[10];
    const float* gamma    = (const float*)d_in[11];
    const float* beta     = (const float*)d_in[12];

    constexpr int Bb = 32, Ss = 256, H = 512;
    constexpr int R  = Bb * Ss;      // 8192 rows
    constexpr int G  = 7 * H;        // 3584
    constexpr int C4 = 4 * H;        // 2048 (h | hb | ha | x)

    char* p = (char*)d_ws;
    auto alloc = [&](size_t bytes) {
        char* r = p;
        p += (bytes + 255) & ~(size_t)255;
        return r;
    };
    __bf16* WgT   = (__bf16*)alloc((size_t)G * C4 * 2);        // [3584][2048]
    __bf16* WtT   = (__bf16*)alloc((size_t)H * H * 2);         // [512][512]
    __bf16* W1T   = (__bf16*)alloc((size_t)(2 * H) * H * 2);   // [1024][512]
    __bf16* W2T   = (__bf16*)alloc((size_t)H * (2 * H) * 2);   // [512][1024]
    __bf16* cat   = (__bf16*)alloc((size_t)R * C4 * 2);
    float*  gates = (float*)alloc((size_t)R * G * 4);
    float*  h     = (float*)alloc((size_t)R * H * 4);
    float*  c0    = (float*)alloc((size_t)R * H * 4);
    float*  c1    = (float*)alloc((size_t)R * H * 4);
    float*  emb   = (float*)alloc((size_t)R * H * 4);
    float*  hn    = (float*)alloc((size_t)R * H * 4);
    __bf16* hnb   = (__bf16*)alloc((size_t)R * H * 2);
    __bf16* ffn1  = (__bf16*)alloc((size_t)R * (2 * H) * 2);

    const int thr = 256;
    // Stacked gate weight: rows 0..3H-1 = Wg, rows 3H..4H-1 = We_gate
    transpose_convert<<<(3 * H * G + thr - 1) / thr, thr, 0, stream>>>(Wg, WgT, 3 * H, G, C4, 0);
    transpose_convert<<<(H * G + thr - 1) / thr, thr, 0, stream>>>(We_gate, WgT, H, G, C4, 3 * H);
    transpose_convert<<<(H * H + thr - 1) / thr, thr, 0, stream>>>(We_trans, WtT, H, H, H, 0);
    transpose_convert<<<(H * 2 * H + thr - 1) / thr, thr, 0, stream>>>(W1, W1T, H, 2 * H, H, 0);
    transpose_convert<<<(2 * H * H + thr - 1) / thr, thr, 0, stream>>>(W2, W2T, 2 * H, H, 2 * H, 0);

    init_state<<<R * H / 256, 256, 0, stream>>>(wi, mask, h, c0, cat);

    // emb_trans = tanh(x @ We_trans + be_trans)   (x = cat block 3)
    gemm_bf16<<<dim3(H / 256, R / 128), 256, 0, stream>>>(
        cat + 3 * H, C4, WtT, H, be_trans, nullptr, 0, emb, nullptr, H, H, 2);

    float* cold = c0;
    float* cnew = c1;
    for (int l = 0; l < 7; ++l) {
        build_cat<<<R * H / 256, 256, 0, stream>>>(h, mask, cat);
        // gates = [h,hb,ha,x] @ [Wg;We_gate] + bg
        gemm_bf16<<<dim3(G / 256, R / 128), 256, 0, stream>>>(
            cat, C4, WgT, C4, bg, nullptr, 0, gates, nullptr, G, C4, 0);
        cell_update<<<R, 256, 0, stream>>>(gates, emb, cold, cnew, h, mask, gamma, beta);
        float* t = cold; cold = cnew; cnew = t;
    }

    final_ln<<<R, 256, 0, stream>>>(h, gamma, beta, hn, hnb);
    // ffn1 = relu(hn @ W1 + b1)  (bf16)
    gemm_bf16<<<dim3(2 * H / 256, R / 128), 256, 0, stream>>>(
        hnb, H, W1T, H, b1, nullptr, 0, nullptr, ffn1, 2 * H, H, 1);
    // out = hn + ffn1 @ W2 + b2
    gemm_bf16<<<dim3(H / 256, R / 128), 256, 0, stream>>>(
        ffn1, 2 * H, W2T, 2 * H, b2, hn, H, (float*)d_out, nullptr, H, 2 * H, 0);
}
